// RLCM_42880953483522
// MI455X (gfx1250) — matrix operand order repaired and checked by reading, not compiled, not run on hardware
//
#include <hip/hip_runtime.h>
#include <hip/hip_bf16.h>

typedef __attribute__((ext_vector_type(16))) __bf16 v16bf;
typedef __attribute__((ext_vector_type(8)))  float  v8f;

#define WMMA_BF16(a,b,c) __builtin_amdgcn_wmma_f32_16x16x32_bf16(false,(a),false,(b),(short)0,(c),false,false)

__device__ inline v8f zero8(){
  v8f z;
#pragma unroll
  for (int i=0;i<8;++i) z[i]=0.f;
  return z;
}

__device__ inline float gelu_exact(float x){
  return 0.5f*x*(1.f+erff(x*0.7071067811865476f));
}

// Fragment-order tiles: each 16x16(x32) operand tile is stored as 512 bf16 =
// [lane 0..31][element 0..15]; a lane's fragment is 32 contiguous bytes.
__device__ inline v16bf frag_ld(const __bf16* p, int lane){
  return *(const v16bf*)(p + lane*16);
}
// A-operand element mapping (16x32): lane = m + 16*half,
// element i <-> klocal: k = i + 16*half.
// Inverse (writer side): half = kl>>4 ; i = kl & 15.
__device__ inline int a_off(int m, int kl){
  int half = kl>>4;
  int i = kl & 15;
  return (m + 16*half)*16 + i;
}

// ---------------- weight packing ----------------
// Conv weights OIHW fp32 -> fragment-ordered bf16 A-tiles, K reordered to (ky,kx,ci).
template<int CIN, int COUT>
__global__ void k_pack_a(const float* __restrict__ w, __bf16* __restrict__ d){
  constexpr int K  = CIN*9;
  constexpr int KB = K/32;
  int idx = blockIdx.x*256 + threadIdx.x;
  if (idx >= COUT*K) return;
  int e    = idx & 15;
  int lane = (idx >> 4) & 31;
  int tile = idx >> 9;
  int kb = tile % KB, mt = tile / KB;
  int m = lane & 15, half = lane >> 4;
  int kl = e + 16*half;
  int k  = kb*32 + kl;
  int co = mt*16 + m;
  int pos = k / CIN, ci = k % CIN;      // k = (ky*3+kx)*CIN + ci
  int ky = pos/3, kx = pos - 3*ky;
  d[idx] = (__bf16)w[((co*CIN + ci)*3 + ky)*3 + kx];
}
// Row-major [K][N] fp32 (or its transpose when TRANS) -> fragment-ordered B-tiles.
// B element mapping (32x16): lane = n + 16*half, element e <-> k = e + 16*half.
template<int K, int N, bool TRANS>
__global__ void k_pack_b(const float* __restrict__ w, __bf16* __restrict__ d){
  constexpr int KB = K/32;
  int idx = blockIdx.x*256 + threadIdx.x;
  if (idx >= K*N) return;
  int e    = idx & 15;
  int lane = (idx >> 4) & 31;
  int tile = idx >> 9;
  int kb = tile % KB, nt = tile / KB;
  int n = nt*16 + (lane & 15);
  int k = kb*32 + e + 16*(lane >> 4);
  d[idx] = (__bf16)(TRANS ? w[n*K + k] : w[k*N + n]);
}

// ---------------- Haar DWT ----------------
__global__ void k_dwt(const float* __restrict__ R, float* __restrict__ ll,
                      float* __restrict__ lh, float* __restrict__ hl, float* __restrict__ hh){
  size_t i = (size_t)blockIdx.x*256 + threadIdx.x;   // 4*64*128*128
  int x = (int)(i & 127), y = (int)((i>>7)&127);
  int c = (int)((i>>14)&63), b = (int)(i>>20);
  const float* p = R + (((size_t)(b*64+c)*256 + 2*y)*256 + 2*x);
  float a = p[0], bb = p[1], cc = p[256], dd = p[257];
  ll[i] = 0.5f*(a+bb+cc+dd);
  lh[i] = 0.5f*(cc+dd-a-bb);
  hl[i] = 0.5f*(bb-a+dd-cc);
  hh[i] = 0.5f*(a-bb-cc+dd);
}

// ---------------- implicit-GEMM 3x3 conv (WMMA bf16) ----------------
// in: NCHW fp32; wt: packed A-tiles (K order (ky,kx,ci)); out: pixel-major fp32.
template<int CIN, int COUT, bool GELU>
__global__ __launch_bounds__(128) void k_conv3x3(const float* __restrict__ in,
    const __bf16* __restrict__ wt, const float* __restrict__ bias,
    float* __restrict__ out, int H, int W){
  constexpr int K  = CIN*9;
  constexpr int KB = K/32;
  constexpr int MT = COUT/16;
  constexpr int PW = 66;                 // 64-wide tile + halo
  extern __shared__ char smem[];
  __bf16* sW = (__bf16*)smem;            // COUT*K (packed tiles)
  __bf16* sP = sW + COUT*K;              // [3][PW][CIN], ci fastest
  int tid = threadIdx.x, lane = tid & 31, wave = tid >> 5;
  int x0 = blockIdx.x*64, y = blockIdx.y, b = blockIdx.z;
  __builtin_prefetch(wt, 0, 1);          // global_prefetch_b8
  for (int i = tid; i < COUT*K; i += 128) sW[i] = wt[i];
  for (int i = tid; i < 3*PW*CIN; i += 128){
    int ci = i % CIN; int col = (i / CIN) % PW; int row = i / (CIN*PW);
    int yy = y + row - 1, xx = x0 + col - 1;
    float val = 0.f;
    if (yy >= 0 && yy < H && xx >= 0 && xx < W)
      val = in[((size_t)(b*CIN + ci)*H + yy)*W + xx];
    sP[i] = (__bf16)val;
  }
  __syncthreads();
  int xl = wave*16;                       // each wave: 16 pixels
  v8f acc[MT];
#pragma unroll
  for (int mt=0; mt<MT; ++mt) acc[mt] = zero8();
  int n = lane & 15, half = lane >> 4;
  for (int kk = 0; kk < K; kk += 32){
    // B fragment: lane's 16 elements are 16 consecutive K = 16 consecutive ci
    int kstart = kk + 16*half;
    int pos = kstart / CIN, ci0 = kstart % CIN;
    int ky = pos/3, kx = pos - 3*ky;
    v16bf bfrag = *(const v16bf*)(sP + (ky*PW + xl + n + kx)*CIN + ci0);
#pragma unroll
    for (int mt=0; mt<MT; ++mt){
      v16bf afrag = frag_ld(sW + (mt*KB + (kk>>5))*512, lane);
      acc[mt] = WMMA_BF16(afrag, bfrag, acc[mt]);
    }
  }
  size_t pix = ((size_t)b*H + y)*W + x0 + xl + n;
  float* orow = out + pix*COUT;
#pragma unroll
  for (int mt=0; mt<MT; ++mt){
    int cb = mt*16 + 8*half;              // 8 consecutive output channels
    float r[8];
#pragma unroll
    for (int v2=0; v2<8; ++v2){
      float val = acc[mt][v2] + bias[cb + v2];
      r[v2] = GELU ? gelu_exact(val) : val;
    }
    *(float4*)(orow + cb)     = make_float4(r[0],r[1],r[2],r[3]);
    *(float4*)(orow + cb + 4) = make_float4(r[4],r[5],r[6],r[7]);
  }
}

// ---------------- offset conv2 + avgpool ----------------
__global__ void k_off2(const float* __restrict__ o1, const float* __restrict__ w2,
                       const float* __restrict__ b2, float* __restrict__ offs){
  __shared__ float sw[6*32*9];
  __shared__ float sb[6];
  for (int i = threadIdx.x; i < 6*32*9; i += blockDim.x) sw[i] = w2[i];
  if (threadIdx.x < 6) sb[threadIdx.x] = b2[threadIdx.x];
  __syncthreads();
  size_t i = (size_t)blockIdx.x*256 + threadIdx.x;     // 4*6*128*128
  int xh = (int)(i & 127), yh = (int)((i>>7)&127);
  int co = (int)((i>>14)%6), b = (int)((i>>14)/6);
  float acc = 0.f;
  for (int dy=0; dy<2; ++dy)
    for (int dx=0; dx<2; ++dx){
      int y = 2*yh+dy, x = 2*xh+dx;
      float s = sb[co];
      for (int ky=0;ky<3;++ky){
        int yy=y+ky-1; if (yy<0||yy>=256) continue;
        for (int kx=0;kx<3;++kx){
          int xx=x+kx-1; if (xx<0||xx>=256) continue;
          const float* px = o1 + ((size_t)(b*256+yy)*256+xx)*32;
          const float* wr = sw + co*288 + ky*3 + kx;
          for (int ci=0; ci<32; ++ci) s += wr[ci*9] * px[ci];
        }
      }
      acc += s;
    }
  offs[i] = 0.25f*acc;
}

// ---------------- deformable bilinear resample ----------------
__device__ inline float reflectf(float ix, float m){
  ix = fabsf(ix);
  ix = fmodf(ix, 2.f*m);
  return ix > m ? 2.f*m - ix : ix;
}
__global__ void k_deform(const float* __restrict__ in, const float* __restrict__ offs,
                         int cb, float* __restrict__ out){
  size_t i = (size_t)blockIdx.x*256 + threadIdx.x;     // 4*64*128*128
  int x = (int)(i & 127), y = (int)((i>>7)&127);
  int c = (int)((i>>14)&63), b = (int)(i>>20);
  const int HW = 16384;
  float o0 = offs[(size_t)(b*6 + cb    )*HW + y*128 + x];
  float o1 = offs[(size_t)(b*6 + cb + 1)*HW + y*128 + x];
  float gx = -1.f + (2.f/127.f)*x + tanhf(o0)*0.25f;
  float gy = -1.f + (2.f/127.f)*y + tanhf(o1)*0.25f;
  float ix = reflectf((gx+1.f)*0.5f*127.f, 127.f);
  float iy = reflectf((gy+1.f)*0.5f*127.f, 127.f);
  float x0f = floorf(ix), y0f = floorf(iy);
  float wx = ix - x0f, wy = iy - y0f;
  int x0 = min(max((int)x0f,0),127), x1 = min(x0+1,127);
  int y0 = min(max((int)y0f,0),127), y1 = min(y0+1,127);
  const float* img = in + (size_t)(b*64+c)*HW;
  float v00=img[y0*128+x0], v01=img[y0*128+x1];
  float v10=img[y1*128+x0], v11=img[y1*128+x1];
  out[i] = v00*(1.f-wx)*(1.f-wy) + v01*wx*(1.f-wy) + v10*(1.f-wx)*wy + v11*wx*wy;
}

// ---------------- LN + FFN + residual (WMMA) ----------------
// z pixel-major [Npix][64] -> fout pixel-major [Npix][64] = z + FFN(LN(z))
__global__ __launch_bounds__(128) void k_ffn(const float* __restrict__ z,
    const float* __restrict__ ln_w, const float* __restrict__ ln_b,
    const __bf16* __restrict__ w1, const float* __restrict__ b1,
    const __bf16* __restrict__ w2, const float* __restrict__ b2,
    float* __restrict__ fout){
  extern __shared__ char smem[];
  float* sZ   = (float*)smem;          // 64*64
  float* sB1  = sZ + 4096;             // 256
  float* sLnW = sB1 + 256;
  float* sLnB = sLnW + 64;
  float* sB2  = sLnB + 64;
  __bf16* sW1 = (__bf16*)(sB2 + 64);   // 16384 : B tiles (nt*2+kb)
  __bf16* sW2 = sW1 + 16384;           // 16384 : B tiles (nt*8+kb)
  __bf16* sX  = sW2 + 16384;           // 4096  : A tiles (wave*2+kb)
  __bf16* sH  = sX + 4096;             // 16384 : A tiles (wave*8+kb)
  int tid = threadIdx.x, lane = tid&31, wave = tid>>5;
  size_t pixBase = (size_t)blockIdx.x * 64;
  __builtin_prefetch(w1, 0, 1);
  for (int i=tid;i<16384;i+=128){ sW1[i]=w1[i]; sW2[i]=w2[i]; }
  for (int i=tid;i<256;i+=128) sB1[i]=b1[i];
  if (tid<64){ sLnW[tid]=ln_w[tid]; sLnB[tid]=ln_b[tid]; sB2[tid]=b2[tid]; }
  __syncthreads();
  if (tid < 64){                        // LayerNorm, one pixel per thread
    const float* zr = z + (pixBase + tid)*64;
    float mu=0.f;
    for (int c=0;c<64;++c) mu += zr[c];
    mu *= (1.f/64.f);
    float var=0.f;
    for (int c=0;c<64;++c){ float d=zr[c]-mu; var += d*d; }
    var *= (1.f/64.f);
    float rs = rsqrtf(var + 1e-5f);
    int mt = tid>>4, m = tid&15;
    for (int c=0;c<64;++c){
      float zv = zr[c];
      sZ[tid*64+c] = zv;
      float xn = (zv-mu)*rs*sLnW[c] + sLnB[c];
      sX[(mt*2 + (c>>5))*512 + a_off(m, c&31)] = (__bf16)xn;   // fragment order
    }
  }
  __syncthreads();
  int n_lo = lane&15, halfC = lane>>4;
  // GEMM1: [16x64]@[64x256] + GELU ; A fragments hoisted
  v16bf ax0 = frag_ld(sX + (wave*2+0)*512, lane);
  v16bf ax1 = frag_ld(sX + (wave*2+1)*512, lane);
  for (int nt=0; nt<16; ++nt){
    v8f acc = zero8();
    acc = WMMA_BF16(ax0, frag_ld(sW1 + (nt*2+0)*512, lane), acc);
    acc = WMMA_BF16(ax1, frag_ld(sW1 + (nt*2+1)*512, lane), acc);
    int k2 = nt*16 + n_lo;              // hidden index = K of GEMM2
#pragma unroll
    for (int v2=0; v2<8; ++v2){
      int m = v2 + 8*halfC;
      float h = gelu_exact(acc[v2] + sB1[k2]);
      sH[(wave*8 + (k2>>5))*512 + a_off(m, k2&31)] = (__bf16)h;  // fragment order
    }
  }
  // GEMM2: [16x256]@[256x64] + residual (wave-private rows; no barrier needed)
  v16bf ah[8];
#pragma unroll
  for (int kb=0;kb<8;++kb) ah[kb] = frag_ld(sH + (wave*8+kb)*512, lane);
  for (int nt=0; nt<4; ++nt){
    v8f acc = zero8();
#pragma unroll
    for (int kb=0;kb<8;++kb)
      acc = WMMA_BF16(ah[kb], frag_ld(sW2 + (nt*8+kb)*512, lane), acc);
    int c = nt*16 + n_lo;
#pragma unroll
    for (int v2=0; v2<8; ++v2){
      int r = (wave*16 + v2 + 8*halfC)*64 + c;
      sZ[r] = acc[v2] + sB2[c] + sZ[r];
    }
  }
  __syncthreads();
  if (tid < 64){                        // coalesced 256B row store per pixel
    const float4* src = (const float4*)(sZ + tid*64);
    float4* dst = (float4*)(fout + (pixBase + tid)*64);
#pragma unroll
    for (int q=0;q<16;++q) dst[q] = src[q];
  }
}

// ---------------- fusion: 1x1 conv (256->64) + LN + FFN + residual -> NCHW ----------------
__global__ __launch_bounds__(128) void k_fusion(const float* __restrict__ f0,
    const float* __restrict__ f1, const float* __restrict__ f2, const float* __restrict__ f3,
    const __bf16* __restrict__ wfT, const float* __restrict__ fus_b,
    const float* __restrict__ ln_w, const float* __restrict__ ln_b,
    const __bf16* __restrict__ w1, const float* __restrict__ b1,
    const __bf16* __restrict__ w2, const float* __restrict__ b2,
    float* __restrict__ dout){
  extern __shared__ char smem[];
  float* sZ   = (float*)smem;           // 64*64
  float* sB1  = sZ + 4096;
  float* sLnW = sB1 + 256;
  float* sLnB = sLnW + 64;
  float* sB2  = sLnB + 64;
  float* sFB  = sB2 + 64;
  __bf16* sW1 = (__bf16*)(sFB + 64);    // 16384
  __bf16* sW2 = sW1 + 16384;            // 16384
  __bf16* sWf = sW2 + 16384;            // 16384 : B tiles (nt*8+kb)
  __bf16* sX  = sWf + 16384;            // 4096  : A tiles (wave*2+kb)
  __bf16* sXc = sX + 4096;              // 16384 : A tiles (wave*8+kb), reused as sH
  __bf16* sH  = sXc;
  int tid=threadIdx.x, lane=tid&31, wave=tid>>5;
  size_t pixBase = (size_t)blockIdx.x*64;
  __builtin_prefetch(wfT, 0, 1);
  for (int i=tid;i<16384;i+=128){ sW1[i]=w1[i]; sW2[i]=w2[i]; sWf[i]=wfT[i]; }
  for (int i=tid;i<256;i+=128) sB1[i]=b1[i];
  if (tid<64){ sLnW[tid]=ln_w[tid]; sLnB[tid]=ln_b[tid]; sB2[tid]=b2[tid]; sFB[tid]=fus_b[tid]; }
  const float* fb[4] = {f0,f1,f2,f3};
  for (int i=tid;i<64*256;i+=128){      // stage concat features in fragment order
    int p = i >> 8, c = i & 255;
    const float* src = fb[c>>6];
    float v = src[(pixBase+p)*64 + (c&63)];
    sXc[((p>>4)*8 + (c>>5))*512 + a_off(p&15, c&31)] = (__bf16)v;
  }
  __syncthreads();
  int n_lo = lane&15, halfC = lane>>4;
  // GEMM0: X[16x256]@WfT[256x64] -> z
  {
    v16bf ac[8];
#pragma unroll
    for (int kb=0;kb<8;++kb) ac[kb] = frag_ld(sXc + (wave*8+kb)*512, lane);
    for (int nt=0; nt<4; ++nt){
      v8f acc = zero8();
#pragma unroll
      for (int kb=0;kb<8;++kb)
        acc = WMMA_BF16(ac[kb], frag_ld(sWf + (nt*8+kb)*512, lane), acc);
      int c = nt*16 + n_lo;
#pragma unroll
      for (int v2=0; v2<8; ++v2)
        sZ[(wave*16 + v2 + 8*halfC)*64 + c] = acc[v2] + sFB[c];
    }
  }
  __syncthreads();
  if (tid < 64){                        // LayerNorm on z
    const float* zr = sZ + tid*64;
    float mu=0.f;
    for (int c=0;c<64;++c) mu += zr[c];
    mu *= (1.f/64.f);
    float var=0.f;
    for (int c=0;c<64;++c){ float d=zr[c]-mu; var+=d*d; }
    var *= (1.f/64.f);
    float rs = rsqrtf(var+1e-5f);
    int mt = tid>>4, m = tid&15;
    for (int c=0;c<64;++c){
      float xn = (zr[c]-mu)*rs*sLnW[c] + sLnB[c];
      sX[(mt*2 + (c>>5))*512 + a_off(m, c&31)] = (__bf16)xn;
    }
  }
  __syncthreads();
  // GEMM1 + GELU (overwrites this wave's own sXc tiles, already consumed)
  v16bf ax0 = frag_ld(sX + (wave*2+0)*512, lane);
  v16bf ax1 = frag_ld(sX + (wave*2+1)*512, lane);
  for (int nt=0; nt<16; ++nt){
    v8f acc = zero8();
    acc = WMMA_BF16(ax0, frag_ld(sW1 + (nt*2+0)*512, lane), acc);
    acc = WMMA_BF16(ax1, frag_ld(sW1 + (nt*2+1)*512, lane), acc);
    int k2 = nt*16 + n_lo;
#pragma unroll
    for (int v2=0; v2<8; ++v2){
      int m = v2 + 8*halfC;
      float h = gelu_exact(acc[v2] + sB1[k2]);
      sH[(wave*8 + (k2>>5))*512 + a_off(m, k2&31)] = (__bf16)h;
    }
  }
  // GEMM2 + residual -> NCHW, two float4 stores per (nt,lane)
  v16bf ah[8];
#pragma unroll
  for (int kb=0;kb<8;++kb) ah[kb] = frag_ld(sH + (wave*8+kb)*512, lane);
  for (int nt=0; nt<4; ++nt){
    v8f acc = zero8();
#pragma unroll
    for (int kb=0;kb<8;++kb)
      acc = WMMA_BF16(ah[kb], frag_ld(sW2 + (nt*8+kb)*512, lane), acc);
    int c = nt*16 + n_lo;
    float r[8];
#pragma unroll
    for (int v2=0; v2<8; ++v2)
      r[v2] = acc[v2] + sB2[c] + sZ[(wave*16 + v2 + 8*halfC)*64 + c];
    size_t pix0 = pixBase + wave*16 + 8*halfC;       // pixels consecutive in v2
    int x0 = (int)(pix0 & 127), yy = (int)((pix0>>7)&127), b = (int)(pix0>>14);
    float* drow = dout + (((size_t)b*64 + c)<<14) + (yy<<7) + x0;
    *(float4*)(drow)     = make_float4(r[0],r[1],r[2],r[3]);
    *(float4*)(drow + 4) = make_float4(r[4],r[5],r[6],r[7]);
  }
}

extern "C" void kernel_launch(void* const* d_in, const int* in_sizes, int n_in,
                              void* d_out, int out_size, void* d_ws, size_t ws_size,
                              hipStream_t stream) {
  const float* R      = (const float*)d_in[0];
  const float* off_w1 = (const float*)d_in[1];
  const float* off_b1 = (const float*)d_in[2];
  const float* off_w2 = (const float*)d_in[3];
  const float* off_b2 = (const float*)d_in[4];
  const float* w_ll   = (const float*)d_in[5];
  const float* b_ll   = (const float*)d_in[6];
  const float* w_lh   = (const float*)d_in[7];
  const float* b_lh   = (const float*)d_in[8];
  const float* w_hl   = (const float*)d_in[9];
  const float* b_hl   = (const float*)d_in[10];
  const float* w_hh   = (const float*)d_in[11];
  const float* b_hh   = (const float*)d_in[12];
  const float* ln_w   = (const float*)d_in[13];
  const float* ln_b   = (const float*)d_in[14];
  const float* ffn_w1 = (const float*)d_in[15];
  const float* ffn_b1 = (const float*)d_in[16];
  const float* ffn_w2 = (const float*)d_in[17];
  const float* ffn_b2 = (const float*)d_in[18];
  const float* fus_w  = (const float*)d_in[19];
  const float* fus_b  = (const float*)d_in[20];

  char* ws = (char*)d_ws;
  const size_t S = (size_t)4*64*128*128*4;   // one half-res fp32 tensor = 16 MB
  float* ll   = (float*)(ws + 0*S);
  float* lh   = (float*)(ws + 1*S);
  float* hl   = (float*)(ws + 2*S);
  float* hh   = (float*)(ws + 3*S);
  float* Zlh  = (float*)(ws + 4*S);
  float* Zhl  = (float*)(ws + 5*S);
  float* Zhh  = (float*)(ws + 6*S);
  float* o1   = (float*)(ws + 7*S);          // [B*256*256][32] = 2S
  float* offs = (float*)(ws + 9*S);          // [4][6][128][128]
  float* z_ll = (float*)(ws + 10*S);
  float* z_lh = (float*)(ws + 11*S);
  float* z_hl = (float*)(ws + 12*S);
  float* z_hh = (float*)(ws + 13*S);
  __bf16* wb      = (__bf16*)(ws + 14*S);
  __bf16* wbf_ll  = wb;
  __bf16* wbf_lh  = wb + 36864;
  __bf16* wbf_hl  = wb + 73728;
  __bf16* wbf_hh  = wb + 110592;
  __bf16* wbf_o1  = wb + 147456;             // 32*576
  __bf16* wbf_f1  = wb + 165888;             // 64*256
  __bf16* wbf_f2  = wb + 182272;             // 256*64
  __bf16* wbf_fT  = wb + 198656;             // 256*64 (transposed fus_w)
  float* f_ll = ll;                          // reuse dead buffers
  float* f_lh = Zlh;
  float* f_hl = Zhl;
  float* f_hh = Zhh;

  // 1. weight packing to fragment-ordered bf16
  k_pack_a<64,64><<<144,256,0,stream>>>(w_ll, wbf_ll);
  k_pack_a<64,64><<<144,256,0,stream>>>(w_lh, wbf_lh);
  k_pack_a<64,64><<<144,256,0,stream>>>(w_hl, wbf_hl);
  k_pack_a<64,64><<<144,256,0,stream>>>(w_hh, wbf_hh);
  k_pack_a<64,32><<<72,256,0,stream>>>(off_w1, wbf_o1);
  k_pack_b<64,256,false><<<64,256,0,stream>>>(ffn_w1, wbf_f1);
  k_pack_b<256,64,false><<<64,256,0,stream>>>(ffn_w2, wbf_f2);
  k_pack_b<256,64,true ><<<64,256,0,stream>>>(fus_w,  wbf_fT);

  // 2. DWT
  k_dwt<<<16384,256,0,stream>>>(R, ll, lh, hl, hh);

  // 3. offset conv1 (64->32, full res) WMMA implicit GEMM + GELU
  size_t lds_c1 = (size_t)(32*576 + 3*66*64)*2;     // 62208 B
  k_conv3x3<64,32,true><<<dim3(4,256,4),128,lds_c1,stream>>>(R, wbf_o1, off_b1, o1, 256, 256);

  // 4. offset conv2 + avgpool (6 ch, memory-bound, scalar)
  k_off2<<<1536,256,0,stream>>>(o1, off_w2, off_b2, offs);

  // 5. deformable resampling of high-freq subbands
  k_deform<<<16384,256,0,stream>>>(lh, offs, 0, Zlh);
  k_deform<<<16384,256,0,stream>>>(hl, offs, 2, Zhl);
  k_deform<<<16384,256,0,stream>>>(hh, offs, 4, Zhh);

  // 6. per-subband 3x3 conv (64->64) WMMA implicit GEMM -> pixel-major z
  size_t lds_c3 = (size_t)(64*576 + 3*66*64)*2;     // 99072 B
  k_conv3x3<64,64,false><<<dim3(2,128,4),128,lds_c3,stream>>>(ll,  wbf_ll, b_ll, z_ll, 128, 128);
  k_conv3x3<64,64,false><<<dim3(2,128,4),128,lds_c3,stream>>>(Zlh, wbf_lh, b_lh, z_lh, 128, 128);
  k_conv3x3<64,64,false><<<dim3(2,128,4),128,lds_c3,stream>>>(Zhl, wbf_hl, b_hl, z_hl, 128, 128);
  k_conv3x3<64,64,false><<<dim3(2,128,4),128,lds_c3,stream>>>(Zhh, wbf_hh, b_hh, z_hh, 128, 128);

  // 7. LN + FFN + residual per subband (WMMA)
  size_t lds_ffn = (size_t)(4096+256+64+64+64)*4 + (size_t)(16384*2+4096+16384)*2; // 124672 B
  k_ffn<<<1024,128,lds_ffn,stream>>>(z_ll, ln_w, ln_b, wbf_f1, ffn_b1, wbf_f2, ffn_b2, f_ll);
  k_ffn<<<1024,128,lds_ffn,stream>>>(z_lh, ln_w, ln_b, wbf_f1, ffn_b1, wbf_f2, ffn_b2, f_lh);
  k_ffn<<<1024,128,lds_ffn,stream>>>(z_hl, ln_w, ln_b, wbf_f1, ffn_b1, wbf_f2, ffn_b2, f_hl);
  k_ffn<<<1024,128,lds_ffn,stream>>>(z_hh, ln_w, ln_b, wbf_f1, ffn_b1, wbf_f2, ffn_b2, f_hh);

  // 8. fusion 1x1 + LN + FFN + residual -> NCHW output
  size_t lds_fus = (size_t)(4096+256+64*4)*4 + (size_t)(16384*3+4096+16384)*2;     // 157696 B
  k_fusion<<<1024,128,lds_fus,stream>>>(f_ll, f_lh, f_hl, f_hh, wbf_fT, fus_b,
                                        ln_w, ln_b, wbf_f1, ffn_b1, wbf_f2, ffn_b2,
                                        (float*)d_out);
}
